// LSTMCellModel1_76424648065291
// MI455X (gfx1250) — matrix operand order repaired and checked by reading, not compiled, not run on hardware
//
#include <hip/hip_runtime.h>
#include <hip/hip_bf16.h>

// ---------------------------------------------------------------------------
// 2-layer LSTM (H=256), batch 65536, 8 recurrent steps, persistent-tile design
// for MI455X (gfx1250, wave32, WMMA f32_16x16x32_f16).
//
// Codegen lessons baked in:
//  - Weight loads are loop-invariant across the 8 recurrent steps; naive code
//    lets LICM hoist all 64 B-fragments -> scratch spills. We make addresses
//    opaque per cell call via an inline-asm zero *offset* (not the pointer,
//    which would lose global address-space provenance and demote the loads to
//    flat_load, coupling LOADcnt with DScnt).
//  - kc loops kept rolled so only a few loads are in flight per wave.
//  - 512 threads (16 waves, 4/SIMD) keeps the full per-wave VGPR budget.
//  - MTILE=32 with two M-subtiles per wave: each B fragment loaded once feeds
//    2 WMMAs, halving the dominant L2 weight traffic (~32 GB for the pass).
// ---------------------------------------------------------------------------

typedef __attribute__((ext_vector_type(16))) _Float16 v16h;
typedef __attribute__((ext_vector_type(8)))  _Float16 v8h;
typedef __attribute__((ext_vector_type(8)))  float    v8f;

#define H        256
#define MTILE    32          // batch rows per workgroup (2 M-subtiles per wave)
#define NTHREADS 512         // 16 waves
#define LS       264         // LDS row stride in halves (528B = 33*16B, bank-conflict free)

__device__ __forceinline__ float sigm_(float v) {
    return 1.0f / (1.0f + __expf(-v));
}
__device__ __forceinline__ float tanh_(float v) {
    // tanh(x) = 1 - 2/(exp(2x)+1)  -> single v_exp_f32 + v_rcp_f32
    return 1.0f - 2.0f / (__expf(2.0f * v) + 1.0f);
}

// Opaque zero (SGPR) so weight-load addresses are not provably invariant
// across the recurrent step loop: blocks LICM/CSE without losing the global
// address space of the base pointer (loads stay global_load_b128).
__device__ __forceinline__ int opaque_zero() {
    int z = 0;
    asm volatile("" : "+s"(z));
    return z;
}

// Load a 16x32 f16 A-fragment from row-major LDS (stride LS halves).
// CDNA5 A layout: lanes 0-15 hold row M=lane, K = kc*32 + {0..7, 16..23};
//                 lanes 16-31 hold row M=lane-16, K = kc*32 + {8..15, 24..31}.
__device__ __forceinline__ v16h load_a_frag(const _Float16* __restrict__ A,
                                            int mbase, int kc, int lane) {
    const int row   = mbase + (lane & 15);
    const int khalf = (lane >> 4) << 3;              // 0 or 8
    const _Float16* p = A + row * LS + kc * 32 + khalf;
    v8h lo = *(const v8h*)(p);                       // K + 0..7   (ds_load_b128)
    v8h hi = *(const v8h*)(p + 16);                  // K + 16..23 (ds_load_b128)
    return __builtin_shufflevector(lo, hi, 0, 1, 2, 3, 4, 5, 6, 7,
                                           8, 9, 10, 11, 12, 13, 14, 15);
}

// One LSTM cell update for this workgroup's 32-row tile.
//  A1/A2 : LDS f16 activations (x-path and h-path), row-major stride LS
//  B1/B2 : fragment-major packed f16 weights ([64 tiles][8 kc][32 lanes][16h])
//  bias  : combined bih+bhh (1024 f32)
//  c     : per-lane cell state c[m*8+r] for the two 16-row M-subtiles
//  Hout  : LDS f16 output hidden state (may alias A2; barrier-protected)
__device__ __forceinline__ void lstm_cell(const _Float16* __restrict__ A1,
                                          const _Float16* __restrict__ A2,
                                          const v16h* __restrict__ B1base,
                                          const v16h* __restrict__ B2base,
                                          const float* __restrict__ bias,
                                          float (&c)[16],
                                          _Float16* __restrict__ Hout,
                                          int lane, int colT) {
    // fresh opaque offset each cell call: addresses differ (as far as the
    // optimizer knows) every recurrent step -> weight loads stay in the loop
    const int zoff = opaque_zero();
    const v16h* B1 = B1base + zoff;
    const v16h* B2 = B2base + zoff;

    const v8f z8 = {0.f, 0.f, 0.f, 0.f, 0.f, 0.f, 0.f, 0.f};
    v8f acc[2][4];
#pragma unroll
    for (int m = 0; m < 2; ++m)
#pragma unroll
        for (int g = 0; g < 4; ++g) acc[m][g] = z8;

    // gates += A1 @ Wih^T   (K = 256 -> 8 chunks of 32)
#pragma clang loop unroll(disable)
    for (int kc = 0; kc < 8; ++kc) {
        v16h a0 = load_a_frag(A1, 0,  kc, lane);
        v16h a1 = load_a_frag(A1, 16, kc, lane);
        const v16h* bp = B1 + (colT * 8 + kc) * 32 + lane;
#pragma unroll
        for (int g = 0; g < 4; ++g) {
            v16h b = bp[g * 4096];   // gate stride = 16 tiles * 8 kc * 32 lanes
            acc[0][g] = __builtin_amdgcn_wmma_f32_16x16x32_f16(
                false, a0, false, b, (short)0, acc[0][g], false, false);
            acc[1][g] = __builtin_amdgcn_wmma_f32_16x16x32_f16(
                false, a1, false, b, (short)0, acc[1][g], false, false);
        }
    }
    // gates += A2 @ Whh^T
#pragma clang loop unroll(disable)
    for (int kc = 0; kc < 8; ++kc) {
        v16h a0 = load_a_frag(A2, 0,  kc, lane);
        v16h a1 = load_a_frag(A2, 16, kc, lane);
        const v16h* bp = B2 + (colT * 8 + kc) * 32 + lane;
#pragma unroll
        for (int g = 0; g < 4; ++g) {
            v16h b = bp[g * 4096];
            acc[0][g] = __builtin_amdgcn_wmma_f32_16x16x32_f16(
                false, a0, false, b, (short)0, acc[0][g], false, false);
            acc[1][g] = __builtin_amdgcn_wmma_f32_16x16x32_f16(
                false, a1, false, b, (short)0, acc[1][g], false, false);
        }
    }

    __syncthreads();  // all waves done reading A2 (Hout may alias it)

    const int n   = lane & 15;           // column within this wave's 16-col tile
    const int ro  = (lane >> 4) << 3;    // row offset within D-tile (0 or 8)
    const int col = colT * 16 + n;       // hidden-unit index
    const float bI = bias[0 * H + col];
    const float bF = bias[1 * H + col];
    const float bG = bias[2 * H + col];
    const float bO = bias[3 * H + col];

#pragma unroll
    for (int m = 0; m < 2; ++m) {
#pragma unroll
        for (int r = 0; r < 8; ++r) {
            float iv = sigm_(acc[m][0][r] + bI);
            float fv = sigm_(acc[m][1][r] + bF);
            float gv = tanh_(acc[m][2][r] + bG);
            float ov = sigm_(acc[m][3][r] + bO);
            float cn = fv * c[m * 8 + r] + iv * gv;
            c[m * 8 + r] = cn;
            float hv = ov * tanh_(cn);
            Hout[(m * 16 + ro + r) * LS + col] = (_Float16)hv;
        }
    }
    __syncthreads();  // h visible to all waves before next cell
}

// ---------------------------------------------------------------------------
// Pre-pack: f32 weights (1024x256 row-major, W so that gates = act @ W^T) into
// fragment-major f16: element e = ((tile*8 + kc)*32 + lane)*16 + j holds
// W[tile*16 + (lane&15)][kc*32 + (lane>>4)*8 + (j&7) + (j>>3)*16].  Also bc = bih + bhh.
// ---------------------------------------------------------------------------
__global__ void pack_weights_kernel(const float* __restrict__ Wih0,
                                    const float* __restrict__ Whh0,
                                    const float* __restrict__ Wih1,
                                    const float* __restrict__ Whh1,
                                    const float* __restrict__ bih0,
                                    const float* __restrict__ bhh0,
                                    const float* __restrict__ bih1,
                                    const float* __restrict__ bhh1,
                                    _Float16* __restrict__ Bout,
                                    float* __restrict__ bc) {
    int idx = blockIdx.x * blockDim.x + threadIdx.x;
    if (idx < 4 * 262144) {
        int mat  = idx >> 18;
        int e    = idx & 262143;
        int j    = e & 15;
        int lane = (e >> 4) & 31;
        int kc   = (e >> 9) & 7;
        int tile = e >> 12;
        int row  = tile * 16 + (lane & 15);
        int col  = kc * 32 + ((lane >> 4) << 3) + (j & 7) + ((j >> 3) << 4);
        const float* src = (mat == 0) ? Wih0 : (mat == 1) ? Whh0
                         : (mat == 2) ? Wih1 : Whh1;
        Bout[idx] = (_Float16)src[row * H + col];
    }
    if (idx < 2048) {
        bc[idx] = (idx < 1024) ? (bih0[idx] + bhh0[idx])
                               : (bih1[idx - 1024] + bhh1[idx - 1024]);
    }
}

// ---------------------------------------------------------------------------
// Main persistent kernel: one workgroup per 32 batch rows, all 8 steps.
// ---------------------------------------------------------------------------
__global__ __launch_bounds__(NTHREADS)
void lstm_persistent_kernel(const float* __restrict__ x,     // (bs,3,2)
                            const float* __restrict__ w0,    // (256,2)
                            const float* __restrict__ b0,    // (256,)
                            const float* __restrict__ w1,    // (2,256)
                            const float* __restrict__ b1,    // (2,)
                            const _Float16* __restrict__ Bw, // packed weights
                            const float* __restrict__ bc,    // combined biases
                            float* __restrict__ out) {       // (bs,12)
    __shared__ _Float16 sX [MTILE * LS];
    __shared__ _Float16 sH1[MTILE * LS];
    __shared__ _Float16 sH2[MTILE * LS];
    __shared__ float    sOut[MTILE * 2];

    const int tid   = threadIdx.x;
    const int lane  = tid & 31;
    const int colT  = tid >> 5;           // wave id = hidden-column tile 0..15
    const int m0    = blockIdx.x * MTILE; // first batch row of this tile

    const v16h* Bih0 = (const v16h*)(Bw);
    const v16h* Bhh0 = (const v16h*)(Bw + 262144);
    const v16h* Bih1 = (const v16h*)(Bw + 524288);
    const v16h* Bhh1 = (const v16h*)(Bw + 786432);
    const float* bc0 = bc;
    const float* bc1 = bc + 1024;

    // zero-init hidden states (cell states live in registers)
    for (int i = tid; i < MTILE * LS; i += NTHREADS) {
        sH1[i] = (_Float16)0.f;
        sH2[i] = (_Float16)0.f;
    }
    float c1[16], c2[16];
#pragma unroll
    for (int i = 0; i < 16; ++i) { c1[i] = 0.f; c2[i] = 0.f; }
    __syncthreads();

    // xe[t] = relu(x[:,t,:] @ w0^T + b0) into sX (f16)
    auto encode_x = [&](int t) {
        for (int i = tid; i < MTILE * H; i += NTHREADS) {
            int m = i >> 8, cidx = i & 255;
            float x0 = x[(m0 + m) * 6 + t * 2 + 0];
            float x1 = x[(m0 + m) * 6 + t * 2 + 1];
            float v  = x0 * w0[cidx * 2] + x1 * w0[cidx * 2 + 1] + b0[cidx];
            sX[m * LS + cidx] = (_Float16)fmaxf(v, 0.0f);
        }
        __syncthreads();
    };

    // xi = relu(out @ w0^T + b0) into sX (f16)
    auto encode_o = [&]() {
        for (int i = tid; i < MTILE * H; i += NTHREADS) {
            int m = i >> 8, cidx = i & 255;
            float v = sOut[m * 2] * w0[cidx * 2]
                    + sOut[m * 2 + 1] * w0[cidx * 2 + 1] + b0[cidx];
            sX[m * LS + cidx] = (_Float16)fmaxf(v, 0.0f);
        }
        __syncthreads();
    };

    // out = h2 @ w1^T + b1 ; write to d_out column pair [2*step, 2*step+1]
    auto project = [&](int step) {
        if (tid < MTILE * 2) {
            int m = tid >> 1, j = tid & 1;
            float s = b1[j];
            const _Float16* hp = &sH2[m * LS];
#pragma unroll
            for (int k = 0; k < H; k += 8) {
                v8h hv = *(const v8h*)(hp + k);
#pragma unroll
                for (int u = 0; u < 8; ++u)
                    s += (float)hv[u] * w1[j * H + k + u];
            }
            sOut[m * 2 + j] = s;
            out[(m0 + m) * 12 + step * 2 + j] = s;
        }
        __syncthreads();
    };

    // ---- 3 warmup steps on encoded waypoints ----
    for (int t = 0; t < 3; ++t) {
        encode_x(t);
        lstm_cell(sX,  sH1, Bih0, Bhh0, bc0, c1, sH1, lane, colT);
        lstm_cell(sH1, sH2, Bih1, Bhh1, bc1, c2, sH2, lane, colT);
    }
    project(0);

    // ---- 5 autoregressive steps ----
    for (int s = 1; s <= 5; ++s) {
        encode_o();
        lstm_cell(sX,  sH1, Bih0, Bhh0, bc0, c1, sH1, lane, colT);
        lstm_cell(sH1, sH2, Bih1, Bhh1, bc1, c2, sH2, lane, colT);
        project(s);
    }
}

// ---------------------------------------------------------------------------
extern "C" void kernel_launch(void* const* d_in, const int* in_sizes, int n_in,
                              void* d_out, int out_size, void* d_ws, size_t ws_size,
                              hipStream_t stream) {
    const float* x    = (const float*)d_in[0];
    const float* w0   = (const float*)d_in[1];
    const float* b0   = (const float*)d_in[2];
    const float* w1   = (const float*)d_in[3];
    const float* b1   = (const float*)d_in[4];
    const float* Wih0 = (const float*)d_in[5];
    const float* Whh0 = (const float*)d_in[6];
    const float* bih0 = (const float*)d_in[7];
    const float* bhh0 = (const float*)d_in[8];
    const float* Wih1 = (const float*)d_in[9];
    const float* Whh1 = (const float*)d_in[10];
    const float* bih1 = (const float*)d_in[11];
    const float* bhh1 = (const float*)d_in[12];

    _Float16* Bw = (_Float16*)d_ws;                           // 4 * 512 KB f16
    float*    bc = (float*)((char*)d_ws + 4u * 262144u * 2u); // 2 MB offset

    pack_weights_kernel<<<4096, 256, 0, stream>>>(
        Wih0, Whh0, Wih1, Whh1, bih0, bhh0, bih1, bhh1, Bw, bc);

    const int bs = in_sizes[0] / 6;  // x has bs*3*2 elements
    lstm_persistent_kernel<<<bs / MTILE, NTHREADS, 0, stream>>>(
        x, w0, b0, w1, b1, Bw, bc, (float*)d_out);
}
